// MNISTBNN_2937757631189
// MI455X (gfx1250) — 8-bit matrix operand order repaired and checked by reading, compiled, not run on hardware
//
#include <hip/hip_runtime.h>
#include <hip/hip_bf16.h>
#include <stdint.h>

typedef int v8i __attribute__((ext_vector_type(8)));

__device__ __forceinline__ int a_off(int v, int h) {
  return ((v >> 2) << 5) | (h << 4) | ((v & 3) << 2);
}
__device__ __forceinline__ int b_off(int v, int h) {
  return ((v >> 2) << 5) | (h << 4) | ((v & 3) << 2);
}

__global__ __launch_bounds__(256) void stem_kernel(
    const float* __restrict__ x, const float* __restrict__ w,
    const float* __restrict__ g, const float* __restrict__ b,
    const float* __restrict__ m, const float* __restrict__ v,
    signed char* __restrict__ act1, int B) {
  __shared__ float sw[288];
  __shared__ float sscale[32], sbias[32];
  int tid = threadIdx.x;
  for (int i = tid; i < 288; i += 256) sw[i] = w[i];
  if (tid < 32) {
    float sc = g[tid] * rsqrtf(v[tid] + 1e-5f);
    sscale[tid] = sc;
    sbias[tid] = b[tid] - m[tid] * sc;
  }
  __syncthreads();
  int p = blockIdx.x * 256 + tid;
  if (p >= B * 784) return;
  int bi = p / 784, yx = p % 784, y = yx / 28, xx = yx % 28;
  const float* xb = x + (size_t)bi * 784;
  float pix[9];
#pragma unroll
  for (int t = 0; t < 9; ++t) {
    int iy = y + t / 3 - 1, ix = xx + t % 3 - 1;
    pix[t] = (iy >= 0 && iy < 28 && ix >= 0 && ix < 28) ? xb[iy * 28 + ix] : 0.f;
  }
  unsigned int* dst = (unsigned int*)(act1 + (size_t)p * 32);
#pragma unroll
  for (int wi = 0; wi < 8; ++wi) {
    unsigned int u = 0;
#pragma unroll
    for (int j = 0; j < 4; ++j) {
      int c = wi * 4 + j;
      float acc = 0.f;
#pragma unroll
      for (int t = 0; t < 9; ++t) acc = fmaf(sw[c * 9 + t], pix[t], acc);
      float f = fmaf(acc, sscale[c], sbias[c]);
      u |= ((f >= 0.f) ? 0x01u : 0xFFu) << (8 * j);
    }
    dst[wi] = u;
  }
}

__global__ void prep_w1(const float* __restrict__ w, signed char* __restrict__ frag) {
  int tid = blockIdx.x * blockDim.x + threadIdx.x;
  if (tid >= 20 * 1024) return;
  int f = tid >> 10, r = tid & 1023;
  int l = r >> 5, v = (r >> 2) & 7, j = r & 3;
  int kt = f >> 2, nt = f & 3;
  int n = nt * 16 + (l & 15), h = l >> 4;
  int k = kt * 64 + b_off(v, h) + j;
  signed char val = 0;
  if (k < 288) {
    int t = k >> 5, ci = k & 31;
    val = (w[(n * 32 + ci) * 9 + t] >= 0.f) ? 1 : -1;
  }
  frag[tid] = val;
}

__global__ void prep_w2(const float* __restrict__ w, signed char* __restrict__ frag) {
  int tid = blockIdx.x * blockDim.x + threadIdx.x;
  if (tid >= 36 * 1024) return;
  int f = tid >> 10, r = tid & 1023;
  int l = r >> 5, v = (r >> 2) & 7, j = r & 3;
  int kt = f >> 2, nt = f & 3;
  int n = nt * 16 + (l & 15), h = l >> 4;
  int k = kt * 64 + b_off(v, h) + j;
  int t = k >> 6, ci = k & 63;
  frag[tid] = (w[(n * 64 + ci) * 9 + t] >= 0.f) ? 1 : -1;
}

__global__ void prep_w3(const float* __restrict__ w, signed char* __restrict__ frag) {
  int tid = blockIdx.x * blockDim.x + threadIdx.x;
  if (tid >= 784 * 1024) return;
  int f = tid >> 10, r = tid & 1023;
  int l = r >> 5, v = (r >> 2) & 7, j = r & 3;
  int kt = f >> 4, nt = f & 15;
  int n = nt * 16 + (l & 15), h = l >> 4;
  int k = kt * 64 + b_off(v, h) + j;
  frag[tid] = (w[(size_t)n * 3136 + k] >= 0.f) ? 1 : -1;
}

__global__ __launch_bounds__(128) void block1_kernel(
    const signed char* __restrict__ act1, const signed char* __restrict__ w1frag,
    const float* __restrict__ g, const float* __restrict__ b,
    const float* __restrict__ m, const float* __restrict__ v,
    signed char* __restrict__ act2) {
  __shared__ signed char smem[930 * 32];
  int bi = blockIdx.x, tid = threadIdx.x;
  unsigned int* s32 = (unsigned int*)smem;
  for (int i = tid; i < 7440; i += 128) s32[i] = 0u;
  __syncthreads();
  const unsigned int* src = (const unsigned int*)(act1 + (size_t)bi * 25088);
  for (int i = tid; i < 6272; i += 128) {
    int row = i / 224, col = i % 224;
    s32[((row + 1) * 30 + 1) * 8 + col] = src[i];
  }
  __syncthreads();

  int lane = tid & 31, nt = tid >> 5;
  int lm = lane & 15, h = lane >> 4;
  int ch = nt * 16 + lm;
  float sc = g[ch] * rsqrtf(v[ch] + 1e-5f);
  float bb = b[ch] - m[ch] * sc;

  v8i bf[5];
#pragma unroll
  for (int kt = 0; kt < 5; ++kt)
    bf[kt] = ((const v8i*)(w1frag + (size_t)(kt * 4 + nt) * 1024))[lane];

#pragma unroll 2
  for (int mt = 0; mt < 49; ++mt) {
    int mrow = mt * 16 + lm;
    int q = mrow >> 2, tp = mrow & 3;
    int oy = 2 * (q / 14) + (tp >> 1), ox = 2 * (q % 14) + (tp & 1);
    v8i acc = {0, 0, 0, 0, 0, 0, 0, 0};
#pragma unroll
    for (int kt = 0; kt < 5; ++kt) {
      v8i a;
#pragma unroll
      for (int vv = 0; vv < 8; ++vv) {
        int k = kt * 64 + a_off(vv, h);
        int t = k >> 5;
        int ci = k & 31;
        a[vv] = *(const int*)(smem + ((oy + t / 3) * 30 + (ox + t % 3)) * 32 + ci);
      }
      acc = __builtin_amdgcn_wmma_i32_16x16x64_iu8(true, a, true, bf[kt], acc, false, false);
    }
    float f0 = fmaxf(fmaxf(fmaf((float)acc[0], sc, bb), fmaf((float)acc[1], sc, bb)),
                     fmaxf(fmaf((float)acc[2], sc, bb), fmaf((float)acc[3], sc, bb)));
    float f1 = fmaxf(fmaxf(fmaf((float)acc[4], sc, bb), fmaf((float)acc[5], sc, bb)),
                     fmaxf(fmaf((float)acc[6], sc, bb), fmaf((float)acc[7], sc, bb)));
    int qa = mt * 4 + 2 * h;
    size_t base = (size_t)bi * 12544;
    act2[base + (size_t)qa * 64 + ch] = (f0 >= 0.f) ? 1 : -1;
    act2[base + (size_t)(qa + 1) * 64 + ch] = (f1 >= 0.f) ? 1 : -1;
  }
}

__global__ __launch_bounds__(128) void block2_kernel(
    const signed char* __restrict__ act2, const signed char* __restrict__ w2frag,
    const float* __restrict__ g, const float* __restrict__ b,
    const float* __restrict__ m, const float* __restrict__ v,
    signed char* __restrict__ act3) {
  __shared__ signed char smem[4 * 16 * 16 * 64];
  int b0 = blockIdx.x * 4, tid = threadIdx.x;
  unsigned int* s32 = (unsigned int*)smem;
  for (int i = tid; i < 16384; i += 128) s32[i] = 0u;
  __syncthreads();
  for (int i = tid; i < 4 * 3136; i += 128) {
    int s = i / 3136, r2 = i % 3136;
    int row = r2 / 224, col = r2 % 224;
    s32[s * 4096 + ((row + 1) * 16 + 1) * 16 + col] =
        ((const unsigned int*)(act2 + (size_t)(b0 + s) * 12544))[r2];
  }
  __syncthreads();

  int lane = tid & 31, nt = tid >> 5;
  int lm = lane & 15, h = lane >> 4;
  int ch = nt * 16 + lm;
  float sc = g[ch] * rsqrtf(v[ch] + 1e-5f);
  float bb = b[ch] - m[ch] * sc;

  v8i bf[9];
#pragma unroll
  for (int kt = 0; kt < 9; ++kt)
    bf[kt] = ((const v8i*)(w2frag + (size_t)(kt * 4 + nt) * 1024))[lane];

#pragma unroll 2
  for (int mt = 0; mt < 49; ++mt) {
    int mrow = mt * 16 + lm;
    int s = mrow / 196, mm = mrow % 196;
    int q = mm >> 2, tp = mm & 3;
    int oy = 2 * (q / 7) + (tp >> 1), ox = 2 * (q % 7) + (tp & 1);
    const signed char* sb = smem + s * 16384;
    v8i acc = {0, 0, 0, 0, 0, 0, 0, 0};
#pragma unroll
    for (int kt = 0; kt < 9; ++kt) {
      const signed char* pr = sb + ((oy + kt / 3) * 16 + (ox + kt % 3)) * 64;
      v8i a;
#pragma unroll
      for (int vv = 0; vv < 8; ++vv) a[vv] = *(const int*)(pr + a_off(vv, h));
      acc = __builtin_amdgcn_wmma_i32_16x16x64_iu8(true, a, true, bf[kt], acc, false, false);
    }
    float f0 = fmaxf(fmaxf(fmaf((float)acc[0], sc, bb), fmaf((float)acc[1], sc, bb)),
                     fmaxf(fmaf((float)acc[2], sc, bb), fmaf((float)acc[3], sc, bb)));
    float f1 = fmaxf(fmaxf(fmaf((float)acc[4], sc, bb), fmaf((float)acc[5], sc, bb)),
                     fmaxf(fmaf((float)acc[6], sc, bb), fmaf((float)acc[7], sc, bb)));
    int quad = mt * 4 + 2 * h;
    int sa = quad / 49, qa = quad % 49;
    int sb2 = (quad + 1) / 49, qb = (quad + 1) % 49;
    act3[(size_t)(b0 + sa) * 3136 + ch * 49 + qa] = (f0 >= 0.f) ? 1 : -1;
    act3[(size_t)(b0 + sb2) * 3136 + ch * 49 + qb] = (f1 >= 0.f) ? 1 : -1;
  }
}

__global__ __launch_bounds__(512) void gemm3_kernel(
    const signed char* __restrict__ act3, const signed char* __restrict__ w3frag,
    const float* __restrict__ g, const float* __restrict__ b,
    const float* __restrict__ m, const float* __restrict__ v,
    float* __restrict__ H) {
  __shared__ signed char smem[16 * 3136];
  int M0 = blockIdx.x * 16, tid = threadIdx.x;
  const unsigned int* src = (const unsigned int*)(act3 + (size_t)M0 * 3136);
  unsigned int* s32 = (unsigned int*)smem;
  for (int i = tid; i < 16 * 784; i += 512) s32[i] = src[i];
  __syncthreads();

  int lane = tid & 31, nt = tid >> 5;
  int lm = lane & 15, h = lane >> 4;
  int ch = nt * 16 + lm;
  float sc = g[ch] * rsqrtf(v[ch] + 1e-5f);
  float bb = b[ch] - m[ch] * sc;

  v8i acc = {0, 0, 0, 0, 0, 0, 0, 0};
  for (int kt = 0; kt < 49; ++kt) {
    if (kt + 4 < 49)
      __builtin_prefetch(w3frag + (size_t)((kt + 4) * 16 + nt) * 1024 + lane * 32, 0, 1);
    const signed char* pr = smem + lm * 3136 + kt * 64;
    v8i a;
#pragma unroll
    for (int vv = 0; vv < 8; ++vv) a[vv] = *(const int*)(pr + a_off(vv, h));
    v8i bfr = ((const v8i*)(w3frag + (size_t)(kt * 16 + nt) * 1024))[lane];
    acc = __builtin_amdgcn_wmma_i32_16x16x64_iu8(true, a, true, bfr, acc, false, false);
  }
#pragma unroll
  for (int r = 0; r < 8; ++r) {
    int mm = M0 + r + 8 * h;
    H[(size_t)mm * 256 + ch] = fmaf((float)acc[r], sc, bb);
  }
}

__global__ __launch_bounds__(256) void fc_kernel(
    const float* __restrict__ H, const float* __restrict__ fw,
    const float* __restrict__ fb, float* __restrict__ out, int B) {
  int t = blockIdx.x * 256 + threadIdx.x;
  if (t >= B * 10) return;
  int bb = t / 10, o = t % 10;
  const float4* hr = (const float4*)(H + (size_t)bb * 256);
  const float4* wr = (const float4*)(fw + o * 256);
  float acc = 0.f;
#pragma unroll 8
  for (int i = 0; i < 64; ++i) {
    float4 hv = hr[i], wv = wr[i];
    acc += hv.x * wv.x + hv.y * wv.y + hv.z * wv.z + hv.w * wv.w;
  }
  out[t] = acc + fb[o];
}

extern "C" void kernel_launch(void* const* d_in, const int* in_sizes, int n_in,
                              void* d_out, int out_size, void* d_ws, size_t ws_size,
                              hipStream_t stream) {
  const float* x = (const float*)d_in[0];
  const float* stem_w = (const float*)d_in[1];
  const float* bn1g = (const float*)d_in[2];
  const float* bn1b = (const float*)d_in[3];
  const float* bn1m = (const float*)d_in[4];
  const float* bn1v = (const float*)d_in[5];
  const float* bn2g = (const float*)d_in[6];
  const float* bn2b = (const float*)d_in[7];
  const float* bn2m = (const float*)d_in[8];
  const float* bn2v = (const float*)d_in[9];
  const float* bn3g = (const float*)d_in[10];
  const float* bn3b = (const float*)d_in[11];
  const float* bn3m = (const float*)d_in[12];
  const float* bn3v = (const float*)d_in[13];
  const float* bn4g = (const float*)d_in[14];
  const float* bn4b = (const float*)d_in[15];
  const float* bn4m = (const float*)d_in[16];
  const float* bn4v = (const float*)d_in[17];
  const float* conv1_w = (const float*)d_in[18];
  const float* conv2_w = (const float*)d_in[19];
  const float* lin_w = (const float*)d_in[20];
  const float* fc_w = (const float*)d_in[21];
  const float* fc_b = (const float*)d_in[22];
  (void)n_in; (void)out_size; (void)ws_size;

  int B = in_sizes[0] / 784;

  char* ws = (char*)d_ws;
  size_t off = 0;
  auto carve = [&](size_t n) { char* p = ws + off; off = (off + n + 1023) & ~(size_t)1023; return p; };
  signed char* act1 = (signed char*)carve((size_t)B * 25088);
  signed char* act2 = (signed char*)carve((size_t)B * 12544);
  signed char* act3 = (signed char*)carve((size_t)B * 3136);
  signed char* w1f  = (signed char*)carve(20 * 1024);
  signed char* w2f  = (signed char*)carve(36 * 1024);
  signed char* w3f  = (signed char*)carve(784 * 1024);
  float*       Hbuf = (float*)carve((size_t)B * 256 * sizeof(float));

  prep_w1<<<(20 * 1024 + 255) / 256, 256, 0, stream>>>(conv1_w, w1f);
  prep_w2<<<(36 * 1024 + 255) / 256, 256, 0, stream>>>(conv2_w, w2f);
  prep_w3<<<(784 * 1024 + 255) / 256, 256, 0, stream>>>(lin_w, w3f);

  stem_kernel<<<(B * 784 + 255) / 256, 256, 0, stream>>>(
      x, stem_w, bn1g, bn1b, bn1m, bn1v, act1, B);

  block1_kernel<<<B, 128, 0, stream>>>(act1, w1f, bn2g, bn2b, bn2m, bn2v, act2);

  block2_kernel<<<B / 4, 128, 0, stream>>>(act2, w2f, bn3g, bn3b, bn3m, bn3v, act3);

  gemm3_kernel<<<B / 16, 512, 0, stream>>>(act3, w3f, bn4g, bn4b, bn4m, bn4v, Hbuf);

  fc_kernel<<<(B * 10 + 255) / 256, 256, 0, stream>>>(Hbuf, fc_w, fc_b, (float*)d_out, B);
}
